// Simple_graph_net_54056458388090
// MI455X (gfx1250) — compile-verified, matrix operand order checked by reading, not run on hardware
//
#include <hip/hip_runtime.h>
#include <hip/hip_bf16.h>
#include <math.h>

typedef __attribute__((ext_vector_type(16))) _Float16 v16h;
typedef __attribute__((ext_vector_type(8)))  _Float16 v8h;
typedef __attribute__((ext_vector_type(8)))  float    v8f;

#define IMG_H   512
#define IMG_W   512
#define NTOK    4096
#define C1      40          // conv channel count
#define KREAL   360         // 9 * 40 real K for conv2 implicit GEMM
#define KT2     12          // K tiles of 32 (384 padded)
#define NT2     3           // N tiles of 16 (48 padded, 40 real)
#define ZIDX    (18 * 18 * C1)   // index of zero pad inside s_act
#define TWO_PI_F 6.28318530717958647692f

// ---------------------------------------------------------------------------
// Pack conv2 weights (3,3,40,40 HWIO f32) into WMMA B-fragment lane order,
// f16, padded to K=384, N=48.  Layout per (ntile, ktile): 32 lanes x 16 halfs.
// B fragment (16-bit, 32x16): col N = lane%16; lanes 0-15 hold K 0-7,16-23;
// lanes 16-31 hold K 8-15,24-31; VGPR v holds pair K = 2v (+8 if v>=4).
// ---------------------------------------------------------------------------
__global__ void pack_b_kernel(const float* __restrict__ w2,
                              _Float16* __restrict__ packedB) {
  int tid = blockIdx.x * blockDim.x + threadIdx.x;
  if (tid >= NT2 * KT2 * 32) return;
  int nt   = tid / (KT2 * 32);
  int rem  = tid % (KT2 * 32);
  int kt   = rem / 32;
  int lane = rem % 32;
  int laneHalf = lane >> 4;
  int co = nt * 16 + (lane & 15);
#pragma unroll
  for (int i = 0; i < 16; ++i) {
    int v = i >> 1, j = i & 1;
    int K = kt * 32 + 2 * v + ((v >= 4) ? 8 : 0) + (laneHalf ? 8 : 0) + j;
    float val = 0.f;
    if (K < KREAL && co < C1) {
      int tap = K / C1, ci = K % C1;            // tap = ky*3+kx
      val = w2[(tap * C1 + ci) * C1 + co];      // HWIO flat
    }
    packedB[(size_t)tid * 16 + i] = (_Float16)val;
  }
}

__global__ void zero_kernel(float* __restrict__ p, int n) {
  int i = blockIdx.x * blockDim.x + threadIdx.x;
  if (i < n) p[i] = 0.f;
}

__device__ __forceinline__ float diff_round(float x) {
  return x - sinf(x * TWO_PI_F) / TWO_PI_F;
}

// ---------------------------------------------------------------------------
// Fused per-token kernel: tokenize+mask, label soft-vote, conv1 (VALU),
// conv2 (WMMA implicit GEMM), dense1 -> 6-dim node features.
// One workgroup (256 thr = 8 wave32) per token.
// ---------------------------------------------------------------------------
__global__ void __launch_bounds__(256)
token_gemm_kernel(const float* __restrict__ img, const float* __restrict__ lab,
                  const float* __restrict__ msk,
                  const float* __restrict__ w1, const float* __restrict__ b1,
                  const float* __restrict__ b2,
                  const float* __restrict__ dw, const float* __restrict__ db,
                  const _Float16* __restrict__ packedB,
                  float* __restrict__ label_sv, float* __restrict__ nodes) {
  __shared__ float    s_in[18 * 18];              // zero-padded masked token
  __shared__ float    s_w1[9 * C1];
  __shared__ float    s_b1[C1];
  __shared__ float    s_b2[48];
  __shared__ alignas(16) _Float16 s_act[18 * 18 * C1 + 8]; // conv1 out + zero pad
  __shared__ _Float16 s_x[256 * C1];              // conv2 output, dense1 input
  __shared__ float    s_red[256];

  const int tid = threadIdx.x;
  const int t   = blockIdx.x;
  const int g   = t >> 10;
  const int idx = t & 1023;
  const int gi = idx >> 5, gj = idx & 31;
  const int sx = g & 1, sy = g >> 1;          // SHIFTS order (0,0)(1,0)(0,1)(1,1)
  const int cellx = 2 * gi + sx, celly = 2 * gj + sy;
  const float cid = (float)(cellx * 64 + celly);

  // stage small weights + zero-fill LDS
  if (tid < 9 * C1) s_w1[tid] = w1[tid];
  if (tid < C1)     s_b1[tid] = b1[tid];
  if (tid < 48)     s_b2[tid] = (tid < C1) ? b2[tid] : 0.f;
  for (int k = tid; k < 18 * 18; k += 256) s_in[k] = 0.f;
  for (int k = tid; k < 18 * 18 * C1 + 8; k += 256) s_act[k] = (_Float16)0.f;
  __syncthreads();

  // ---- tokenize: one pixel per thread ----
  const int p = tid >> 4, q = tid & 15;
  int xi = cellx * 8 - 4 + p;
  int yi = celly * 8 - 4 + q;
  float imv = 0.f, lbv = 0.f, mv = -1.f;
  if (xi >= 0 && xi < IMG_H && yi >= 0 && yi < IMG_W) {
    imv = img[xi * IMG_W + yi];
    lbv = lab[xi * IMG_W + yi];
    mv  = msk[xi * IMG_W + yi];
  }
  float filt = (mv == cid) ? 1.f : 0.f;
  s_in[(p + 1) * 18 + (q + 1)] = imv * filt;

  // ---- label soft-vote reduction ----
  s_red[tid] = filt;
  __syncthreads();
  for (int s = 128; s > 0; s >>= 1) {
    if (tid < s) s_red[tid] += s_red[tid + s];
    __syncthreads();
  }
  float fsum = s_red[0];
  __syncthreads();
  s_red[tid] = lbv * filt;
  __syncthreads();
  for (int s = 128; s > 0; s >>= 1) {
    if (tid < s) s_red[tid] += s_red[tid + s];
    __syncthreads();
  }
  if (tid == 0) {
    float sv = s_red[0] / (fsum + 1e-8f);
    sv = diff_round(diff_round(sv));
    label_sv[t] = sv;
  }
  __syncthreads();

  // ---- conv1 (3x3x1 -> 40), VALU, one output position per thread ----
  {
    float acc[C1];
#pragma unroll
    for (int co = 0; co < C1; ++co) acc[co] = s_b1[co];
#pragma unroll
    for (int dy = 0; dy < 3; ++dy)
#pragma unroll
      for (int dx = 0; dx < 3; ++dx) {
        float v = s_in[(p + dy) * 18 + (q + dx)];
        const float* wr = &s_w1[(dy * 3 + dx) * C1];
#pragma unroll
        for (int co = 0; co < C1; ++co) acc[co] += v * wr[co];
      }
    _Float16* dst = &s_act[((p + 1) * 18 + (q + 1)) * C1];
#pragma unroll
    for (int co = 0; co < C1; ++co) dst[co] = (_Float16)fmaxf(acc[co], 0.f);
  }
  __syncthreads();

  // ---- conv2 as implicit GEMM on WMMA: M=256 pos, N=48(40), K=384(360) ----
  // A fragment (16-bit 16x32): row M = lane%16; lane's 16 halfs are two runs
  // of 8 consecutive K (start K = kt*32 + 8*laneHalf, and +16).  Since the
  // im2col K order is tap*40+c and 40%8==0, each 8-run is 8 contiguous,
  // 16B-aligned f16 in s_act -> exactly 2 x ds_load_b128 per fragment.
  const int lane = tid & 31;
  const int wv = tid >> 5;                // wave 0..7
  const int laneHalf = lane >> 4;
  const int mlocal = lane & 15;

#pragma unroll
  for (int mh = 0; mh < 2; ++mh) {
    const int mt = wv + 8 * mh;                 // position tile: y = mt, x = mlocal
    const int rowBase = (mt * 18 + mlocal) * C1; // halfs
    union { v8f v; float f[8]; } c0, c1, c2;
    c0.v = (v8f){0.f, 0.f, 0.f, 0.f, 0.f, 0.f, 0.f, 0.f};
    c1.v = c0.v;
    c2.v = c0.v;
#pragma unroll
    for (int kt = 0; kt < KT2; ++kt) {
      union { v16h v; v8h r[2]; } a;
#pragma unroll
      for (int run = 0; run < 2; ++run) {
        const int Kb = kt * 32 + 16 * run;      // compile-time after unroll
        // laneHalf==0 -> K = Kb ; laneHalf==1 -> K = Kb + 8
        const int tap0 = Kb / C1,       cc0 = Kb % C1;
        const int tap1 = (Kb + 8) / C1, cc1 = (Kb + 8) % C1;
        const int off0 = ((tap0 / 3) * 18 + (tap0 % 3)) * C1 + cc0;
        const int off1 = ((tap1 / 3) * 18 + (tap1 % 3)) * C1 + cc1;
        const int idx0 = (Kb < KREAL)     ? rowBase + off0 : ZIDX;
        const int idx1 = (Kb + 8 < KREAL) ? rowBase + off1 : ZIDX;
        const int srci = laneHalf ? idx1 : idx0;
        a.r[run] = *(const v8h*)&s_act[srci];
      }
      const v16h b0  = *(const v16h*)(packedB + ((size_t)(0 * KT2 + kt) * 32 + lane) * 16);
      const v16h b1f = *(const v16h*)(packedB + ((size_t)(1 * KT2 + kt) * 32 + lane) * 16);
      const v16h b2f = *(const v16h*)(packedB + ((size_t)(2 * KT2 + kt) * 32 + lane) * 16);
      c0.v = __builtin_amdgcn_wmma_f32_16x16x32_f16(false, a.v, false, b0,  (short)0, c0.v, false, false);
      c1.v = __builtin_amdgcn_wmma_f32_16x16x32_f16(false, a.v, false, b1f, (short)0, c1.v, false, false);
      c2.v = __builtin_amdgcn_wmma_f32_16x16x32_f16(false, a.v, false, b2f, (short)0, c2.v, false, false);
    }
    // C/D layout: VGPR r -> M = r + 8*laneHalf, N = lane%16
#pragma unroll
    for (int r = 0; r < 8; ++r) {
      int M = mt * 16 + r + 8 * laneHalf;
#pragma unroll
      for (int nt = 0; nt < NT2; ++nt) {
        int N = nt * 16 + mlocal;
        if (N < C1) {
          float cv = (nt == 0) ? c0.f[r] : (nt == 1) ? c1.f[r] : c2.f[r];
          s_x[M * C1 + N] = (_Float16)fmaxf(cv + s_b2[N], 0.f);
        }
      }
    }
  }
  __syncthreads();

  // ---- dense1: 10240 -> 6, strided partials + LDS tree reduce ----
  float acc6[6] = {0.f, 0.f, 0.f, 0.f, 0.f, 0.f};
  for (int i = tid; i < 256 * C1; i += 256) {
    float xv = (float)s_x[i];
    const float* wr = &dw[(size_t)i * 6];
#pragma unroll
    for (int o = 0; o < 6; ++o) acc6[o] += xv * wr[o];
  }
  for (int o = 0; o < 6; ++o) {
    __syncthreads();
    s_red[tid] = acc6[o];
    __syncthreads();
    for (int s = 128; s > 0; s >>= 1) {
      if (tid < s) s_red[tid] += s_red[tid + s];
      __syncthreads();
    }
    if (tid == 0) nodes[t * 6 + o] = fmaxf(s_red[0] + db[o], 0.f);
  }
}

// ---------------------------------------------------------------------------
// Edge MLP (13->5->5->5->10) + segment_sum via atomicAdd.
// ---------------------------------------------------------------------------
__global__ void edge_kernel(const float* __restrict__ nodes,
                            const int* __restrict__ snd, const int* __restrict__ rcv,
                            const float* we0, const float* be0,
                            const float* we1, const float* be1,
                            const float* we2, const float* be2,
                            const float* we3, const float* be3,
                            float* __restrict__ agg, int E) {
  int e = blockIdx.x * blockDim.x + threadIdx.x;
  if (e >= E) return;
  int s = snd[e], r = rcv[e];
  float in[13];
#pragma unroll
  for (int k = 0; k < 6; ++k) in[k] = nodes[s * 6 + k];
#pragma unroll
  for (int k = 0; k < 6; ++k) in[6 + k] = nodes[r * 6 + k];
  in[12] = 1.f;
  float h1[5], h2[5], h3[5];
#pragma unroll
  for (int j = 0; j < 5; ++j) {
    float a = be0[j];
    for (int i = 0; i < 13; ++i) a += in[i] * we0[i * 5 + j];
    h1[j] = fmaxf(a, 0.f);
  }
#pragma unroll
  for (int j = 0; j < 5; ++j) {
    float a = be1[j];
    for (int i = 0; i < 5; ++i) a += h1[i] * we1[i * 5 + j];
    h2[j] = fmaxf(a, 0.f);
  }
#pragma unroll
  for (int j = 0; j < 5; ++j) {
    float a = be2[j];
    for (int i = 0; i < 5; ++i) a += h2[i] * we2[i * 5 + j];
    h3[j] = fmaxf(a, 0.f);
  }
#pragma unroll
  for (int j = 0; j < 10; ++j) {
    float a = be3[j];
    for (int i = 0; i < 5; ++i) a += h3[i] * we3[i * 5 + j];
    atomicAdd(&agg[r * 10 + j], a);
  }
}

// ---------------------------------------------------------------------------
// Node MLP (17->5->5->5->10) -> logits -> per-node loss.
// ---------------------------------------------------------------------------
__global__ void node_kernel(const float* __restrict__ nodes,
                            const float* __restrict__ agg,
                            const float* wn0, const float* bn0,
                            const float* wn1, const float* bn1,
                            const float* wn2, const float* bn2,
                            const float* wn3, const float* bn3,
                            const float* wout, const float* bout,
                            const float* __restrict__ label_sv,
                            float* __restrict__ out, int N) {
  int t = blockIdx.x * blockDim.x + threadIdx.x;
  if (t >= N) return;
  float in[17];
#pragma unroll
  for (int k = 0; k < 6; ++k) in[k] = nodes[t * 6 + k];
#pragma unroll
  for (int k = 0; k < 10; ++k) in[6 + k] = agg[t * 10 + k];
  in[16] = 1.f;
  float h1[5], h2[5], h3[5], nn[10];
#pragma unroll
  for (int j = 0; j < 5; ++j) {
    float a = bn0[j];
    for (int i = 0; i < 17; ++i) a += in[i] * wn0[i * 5 + j];
    h1[j] = fmaxf(a, 0.f);
  }
#pragma unroll
  for (int j = 0; j < 5; ++j) {
    float a = bn1[j];
    for (int i = 0; i < 5; ++i) a += h1[i] * wn1[i * 5 + j];
    h2[j] = fmaxf(a, 0.f);
  }
#pragma unroll
  for (int j = 0; j < 5; ++j) {
    float a = bn2[j];
    for (int i = 0; i < 5; ++i) a += h2[i] * wn2[i * 5 + j];
    h3[j] = fmaxf(a, 0.f);
  }
#pragma unroll
  for (int j = 0; j < 10; ++j) {
    float a = bn3[j];
    for (int i = 0; i < 5; ++i) a += h3[i] * wn3[i * 5 + j];
    nn[j] = a;
  }
  float l0 = bout[0], l1 = bout[1];
#pragma unroll
  for (int i = 0; i < 10; ++i) {
    l0 += nn[i] * wout[i * 2 + 0];
    l1 += nn[i] * wout[i * 2 + 1];
  }
  float m = fmaxf(l0, l1);
  float lse = m + logf(expf(l0 - m) + expf(l1 - m));
  float sv = label_sv[t];
  out[t] = -((1.f - sv) * (l0 - lse) + sv * (l1 - lse));
}

// ---------------------------------------------------------------------------
extern "C" void kernel_launch(void* const* d_in, const int* in_sizes, int n_in,
                              void* d_out, int out_size, void* d_ws, size_t ws_size,
                              hipStream_t stream) {
  const float* img  = (const float*)d_in[0];
  const float* lab  = (const float*)d_in[1];
  const float* msk  = (const float*)d_in[2];
  const float* c1w  = (const float*)d_in[3];
  const float* c1b  = (const float*)d_in[4];
  const float* c2w  = (const float*)d_in[5];
  const float* c2b  = (const float*)d_in[6];
  const float* d1w  = (const float*)d_in[7];
  const float* d1b  = (const float*)d_in[8];
  const float* we0  = (const float*)d_in[9];
  const float* be0  = (const float*)d_in[10];
  const float* we1  = (const float*)d_in[11];
  const float* be1  = (const float*)d_in[12];
  const float* we2  = (const float*)d_in[13];
  const float* be2  = (const float*)d_in[14];
  const float* we3  = (const float*)d_in[15];
  const float* be3  = (const float*)d_in[16];
  const float* wn0  = (const float*)d_in[17];
  const float* bn0  = (const float*)d_in[18];
  const float* wn1  = (const float*)d_in[19];
  const float* bn1  = (const float*)d_in[20];
  const float* wn2  = (const float*)d_in[21];
  const float* bn2  = (const float*)d_in[22];
  const float* wn3  = (const float*)d_in[23];
  const float* bn3  = (const float*)d_in[24];
  const float* wout = (const float*)d_in[25];
  const float* bout = (const float*)d_in[26];
  const int*   snd  = (const int*)d_in[27];
  const int*   rcv  = (const int*)d_in[28];
  const int E = in_sizes[27];

  char* ws = (char*)d_ws;
  _Float16* packedB = (_Float16*)(ws);                    // 36864 B
  float* label_sv   = (float*)(ws + 36864);               // 16384 B
  float* nodes      = (float*)(ws + 53248);               // 98304 B
  float* agg        = (float*)(ws + 151552);              // 163840 B

  pack_b_kernel<<<(NT2 * KT2 * 32 + 255) / 256, 256, 0, stream>>>(c2w, packedB);
  zero_kernel<<<(NTOK * 10 + 255) / 256, 256, 0, stream>>>(agg, NTOK * 10);
  token_gemm_kernel<<<NTOK, 256, 0, stream>>>(img, lab, msk, c1w, c1b, c2b,
                                              d1w, d1b, packedB, label_sv, nodes);
  edge_kernel<<<(E + 255) / 256, 256, 0, stream>>>(nodes, snd, rcv,
                                                   we0, be0, we1, be1, we2, be2,
                                                   we3, be3, agg, E);
  node_kernel<<<(NTOK + 255) / 256, 256, 0, stream>>>(nodes, agg,
                                                      wn0, bn0, wn1, bn1,
                                                      wn2, bn2, wn3, bn3,
                                                      wout, bout, label_sv,
                                                      (float*)d_out, NTOK);
}
